// SBERTaModel_6090263625875
// MI455X (gfx1250) — matrix operand order repaired and checked by reading, not compiled, not run on hardware
//
#include <hip/hip_runtime.h>
#include <hip/hip_bf16.h>

#define BB 8
#define TT 512
#define BT (BB*TT)
#define DD 768
#define HH 12
#define DHD 64
#define FF 3072
#define LL 12

typedef __attribute__((ext_vector_type(16))) __bf16 bf16x16;
typedef __attribute__((ext_vector_type(8)))  __bf16 bf16x8;
typedef __attribute__((ext_vector_type(8)))  float  f32x8;

#define WMMA_BF16(a,b,c) __builtin_amdgcn_wmma_f32_16x16x32_bf16(false,(a),false,(b),(short)0,(c),false,false)

// ---------------------------------------------------------------------------
// Fragment loaders (CDNA5 WMMA 16x16x32 bf16 layouts, ISA 7.12.2)
// A (16x32, MxK): lane L holds row M=L%16; half h=L/16.
//   elems 0..7  -> K = k0 + 8h + j       (VGPRs 0-3)
//   elems 8..15 -> K = k0 + 16 + 8h + j  (VGPRs 4-7)
// B (32x16, KxN): lane L holds col N=L%16; elems j use A's K order (k0+8h+j, k0+16+8h+j)
// C/D (16x16 f32): lane L col N=L%16; VGPR v row M = v + 8*(L/16)
// ---------------------------------------------------------------------------
static __device__ __forceinline__ bf16x16 load_a_bf16(const __bf16* __restrict__ A, int lda,
                                                      int m0, int k0, int lane) {
  const __bf16* ap = A + (size_t)(m0 + (lane & 15)) * lda + (k0 + 8 * (lane >> 4));
  bf16x8 lo = *(const bf16x8*)ap;         // 16B aligned
  bf16x8 hi = *(const bf16x8*)(ap + 16);  // 16B aligned
  return __builtin_shufflevector(lo, hi, 0,1,2,3,4,5,6,7,8,9,10,11,12,13,14,15);
}

static __device__ __forceinline__ bf16x16 load_b_bf16(const __bf16* __restrict__ Bt, int ldb,
                                                      int k0, int n0, int lane) {
  // Bt is K-major (K x N) bf16; lane holds column n0+L%16, K in A's table order (strided loads)
  const __bf16* bp = Bt + (size_t)(k0 + 8 * (lane >> 4)) * ldb + n0 + (lane & 15);
  bf16x16 r;
#pragma unroll
  for (int j = 0; j < 8; ++j) {
    r[j]     = bp[(size_t)j * ldb];
    r[8 + j] = bp[(size_t)(16 + j) * ldb];
  }
  return r;
}

template<bool GELU>
static __device__ __forceinline__ void store_frag_f32(float* __restrict__ C, int ldc,
                                                      int m0, int n0, int lane,
                                                      const f32x8& acc,
                                                      const float* __restrict__ bias, float scale) {
  const int n  = n0 + (lane & 15);
  const int mb = m0 + ((lane >> 4) << 3);
  const float bval = bias ? bias[n] : 0.0f;
#pragma unroll
  for (int vv = 0; vv < 8; ++vv) {
    float val = acc[vv] * scale + bval;
    if (GELU) val = 0.5f * val * (1.0f + erff(val * 0.70710678118654752f));
    C[(size_t)(mb + vv) * ldc + n] = val;
  }
}

template<bool GELU>
static __device__ __forceinline__ void store_frag_bf16(__bf16* __restrict__ C, int ldc,
                                                       int m0, int n0, int lane,
                                                       const f32x8& acc,
                                                       const float* __restrict__ bias, float scale) {
  const int n  = n0 + (lane & 15);
  const int mb = m0 + ((lane >> 4) << 3);
  const float bval = bias ? bias[n] : 0.0f;
#pragma unroll
  for (int vv = 0; vv < 8; ++vv) {
    float val = acc[vv] * scale + bval;
    if (GELU) val = 0.5f * val * (1.0f + erff(val * 0.70710678118654752f));
    C[(size_t)(mb + vv) * ldc + n] = (__bf16)val;
  }
}

// K-projection epilogue: write transposed per-head layout kt[(b*H+h)*DH+dh][t]
// so scores_gemm reads it as a K-major B (rows dh, cols t).
static __device__ __forceinline__ void store_frag_kt(__bf16* __restrict__ kt,
                                                     int m0, int n0, int lane,
                                                     const f32x8& acc) {
  const int n  = n0 + (lane & 15);           // global feature 0..D-1
  const int mb = m0 + ((lane >> 4) << 3);    // global token
  const int b = mb / TT, t0 = mb % TT;
  const int h = n / DHD, dh = n % DHD;
  __bf16* dst = kt + ((size_t)(b * HH + h) * DHD + dh) * TT + t0;
#pragma unroll
  for (int vv = 0; vv < 8; ++vv) dst[vv] = (__bf16)acc[vv]; // 8 contiguous t
}

// ---------------------------------------------------------------------------
// GEMM: C(MxN) = A(MxK, bf16) * Wt(KxN, bf16)  [+bias][+exact GELU]
// 64x64 tile per 128-thread block; 4 waves in 2x2; 4 WMMAs per wave per K-step
// MODE: 0 = f32 out, 1 = bf16 out, 2 = bf16 per-head-transposed (K cache)
// ---------------------------------------------------------------------------
template<bool GELU, int MODE>
__global__ __launch_bounds__(128) void gemm_proj(const __bf16* __restrict__ A,
                                                 const __bf16* __restrict__ Wt,
                                                 const float* __restrict__ bias,
                                                 void* __restrict__ Cout,
                                                 int M, int N, int Kd) {
  const int lane = threadIdx.x & 31;
  const int wave = threadIdx.x >> 5;
  const int m0 = blockIdx.y * 64 + (wave >> 1) * 32;
  const int n0 = blockIdx.x * 64 + (wave & 1) * 32;
  f32x8 acc00 = {}, acc01 = {}, acc10 = {}, acc11 = {};
  for (int k0 = 0; k0 < Kd; k0 += 32) {
    if (k0 + 32 < Kd)
      __builtin_prefetch(Wt + (size_t)(k0 + 32 + lane) * N + n0, 0, 1);
    bf16x16 a0 = load_a_bf16(A, Kd, m0,      k0, lane);
    bf16x16 a1 = load_a_bf16(A, Kd, m0 + 16, k0, lane);
    bf16x16 b0 = load_b_bf16(Wt, N, k0, n0,      lane);
    bf16x16 b1 = load_b_bf16(Wt, N, k0, n0 + 16, lane);
    acc00 = WMMA_BF16(a0, b0, acc00);
    acc01 = WMMA_BF16(a0, b1, acc01);
    acc10 = WMMA_BF16(a1, b0, acc10);
    acc11 = WMMA_BF16(a1, b1, acc11);
  }
  if (MODE == 0) {
    float* C = (float*)Cout;
    store_frag_f32<GELU>(C, N, m0,      n0,      lane, acc00, bias, 1.0f);
    store_frag_f32<GELU>(C, N, m0,      n0 + 16, lane, acc01, bias, 1.0f);
    store_frag_f32<GELU>(C, N, m0 + 16, n0,      lane, acc10, bias, 1.0f);
    store_frag_f32<GELU>(C, N, m0 + 16, n0 + 16, lane, acc11, bias, 1.0f);
  } else if (MODE == 1) {
    __bf16* C = (__bf16*)Cout;
    store_frag_bf16<GELU>(C, N, m0,      n0,      lane, acc00, bias, 1.0f);
    store_frag_bf16<GELU>(C, N, m0,      n0 + 16, lane, acc01, bias, 1.0f);
    store_frag_bf16<GELU>(C, N, m0 + 16, n0,      lane, acc10, bias, 1.0f);
    store_frag_bf16<GELU>(C, N, m0 + 16, n0 + 16, lane, acc11, bias, 1.0f);
  } else {
    __bf16* C = (__bf16*)Cout;
    store_frag_kt(C, m0,      n0,      lane, acc00);
    store_frag_kt(C, m0,      n0 + 16, lane, acc01);
    store_frag_kt(C, m0 + 16, n0,      lane, acc10);
    store_frag_kt(C, m0 + 16, n0 + 16, lane, acc11);
  }
}

// scores[b,h] = (q_bh @ k_bh^T) * 1/sqrt(DH); kt is per-head K-major -> strided B loads
__global__ __launch_bounds__(128) void scores_gemm(const __bf16* __restrict__ q,
                                                   const __bf16* __restrict__ kt,
                                                   float* __restrict__ scores) {
  const int lane = threadIdx.x & 31;
  const int wave = threadIdx.x >> 5;
  const int z = blockIdx.z, b = z / HH, h = z % HH;
  const __bf16* A  = q  + (size_t)b * TT * DD + h * DHD;
  const __bf16* Bt = kt + (size_t)z * DHD * TT;      // rows: dh (K), cols: t (N)
  float* C = scores + (size_t)z * TT * TT;
  const int m0 = blockIdx.y * 64 + (wave >> 1) * 32;
  const int n0 = blockIdx.x * 64 + (wave & 1) * 32;
  f32x8 acc00 = {}, acc01 = {}, acc10 = {}, acc11 = {};
#pragma unroll
  for (int k0 = 0; k0 < DHD; k0 += 32) {
    bf16x16 a0 = load_a_bf16(A, DD, m0,      k0, lane);
    bf16x16 a1 = load_a_bf16(A, DD, m0 + 16, k0, lane);
    bf16x16 b0 = load_b_bf16(Bt, TT, k0, n0,      lane);
    bf16x16 b1 = load_b_bf16(Bt, TT, k0, n0 + 16, lane);
    acc00 = WMMA_BF16(a0, b0, acc00);
    acc01 = WMMA_BF16(a0, b1, acc01);
    acc10 = WMMA_BF16(a1, b0, acc10);
    acc11 = WMMA_BF16(a1, b1, acc11);
  }
  const float sc = 0.125f; // 1/sqrt(64)
  store_frag_f32<false>(C, TT, m0,      n0,      lane, acc00, nullptr, sc);
  store_frag_f32<false>(C, TT, m0,      n0 + 16, lane, acc01, nullptr, sc);
  store_frag_f32<false>(C, TT, m0 + 16, n0,      lane, acc10, nullptr, sc);
  store_frag_f32<false>(C, TT, m0 + 16, n0 + 16, lane, acc11, nullptr, sc);
}

// ctx[b,h] = attn_bh (T x T, bf16) @ v_bh (T x DH, bf16) -> bf16 (B,T,D) head slice
__global__ __launch_bounds__(128) void ctx_gemm(const __bf16* __restrict__ attn,
                                                const __bf16* __restrict__ v,
                                                __bf16* __restrict__ ctx) {
  const int lane = threadIdx.x & 31;
  const int wave = threadIdx.x >> 5;
  const int z = blockIdx.z, b = z / HH, h = z % HH;
  const __bf16* A  = attn + (size_t)z * TT * TT;
  const __bf16* Bm = v   + (size_t)b * TT * DD + h * DHD;  // K-major rows (t), cols dh
  __bf16* C = ctx + (size_t)b * TT * DD + h * DHD;
  const int m0 = blockIdx.y * 64 + (wave >> 1) * 32;
  const int n0 = (wave & 1) * 32;                          // N = 64, grid.x = 1
  f32x8 acc00 = {}, acc01 = {}, acc10 = {}, acc11 = {};
  for (int k0 = 0; k0 < TT; k0 += 32) {
    bf16x16 a0 = load_a_bf16(A, TT, m0,      k0, lane);
    bf16x16 a1 = load_a_bf16(A, TT, m0 + 16, k0, lane);
    bf16x16 b0 = load_b_bf16(Bm, DD, k0, n0,      lane);
    bf16x16 b1 = load_b_bf16(Bm, DD, k0, n0 + 16, lane);
    acc00 = WMMA_BF16(a0, b0, acc00);
    acc01 = WMMA_BF16(a0, b1, acc01);
    acc10 = WMMA_BF16(a1, b0, acc10);
    acc11 = WMMA_BF16(a1, b1, acc11);
  }
  store_frag_bf16<false>(C, DD, m0,      n0,      lane, acc00, nullptr, 1.0f);
  store_frag_bf16<false>(C, DD, m0,      n0 + 16, lane, acc01, nullptr, 1.0f);
  store_frag_bf16<false>(C, DD, m0 + 16, n0,      lane, acc10, nullptr, 1.0f);
  store_frag_bf16<false>(C, DD, m0 + 16, n0 + 16, lane, acc11, nullptr, 1.0f);
}

// ---------------------------------------------------------------------------
// Weight transpose + f32->bf16: W (N_ x K_) -> Wt (K_ x N_); one matrix per z
// ---------------------------------------------------------------------------
__global__ __launch_bounds__(256) void wt_transpose(const float* __restrict__ W,
                                                    __bf16* __restrict__ Wt,
                                                    int N_, int K_) {
  __shared__ float tile[32][33];
  const float* Wm  = W  + (size_t)blockIdx.z * N_ * K_;
  __bf16*      Wtm = Wt + (size_t)blockIdx.z * N_ * K_;
  const int k0 = blockIdx.x * 32, n0 = blockIdx.y * 32;
  for (int r = threadIdx.y; r < 32; r += 8)
    tile[r][threadIdx.x] = Wm[(size_t)(n0 + r) * K_ + k0 + threadIdx.x];
  __syncthreads();
  for (int r = threadIdx.y; r < 32; r += 8)
    Wtm[(size_t)(k0 + r) * N_ + n0 + threadIdx.x] = (__bf16)tile[threadIdx.x][r];
}

// ---------------------------------------------------------------------------
// Row LayerNorm: out = LN(x [+ res]) * w + b (f32) plus optional bf16 mirror
// ---------------------------------------------------------------------------
__global__ __launch_bounds__(256) void ln_kernel(const float* __restrict__ x,
                                                 const float* __restrict__ res,
                                                 const float* __restrict__ w,
                                                 const float* __restrict__ bia,
                                                 float* __restrict__ out,
                                                 __bf16* __restrict__ out_bf) {
  __shared__ float red[256];
  const size_t base = (size_t)blockIdx.x * DD;
  float v[3]; float lsum = 0.0f;
#pragma unroll
  for (int i = 0; i < 3; ++i) {
    const int d = threadIdx.x + i * 256;
    float t = x[base + d];
    if (res) t += res[base + d];
    v[i] = t; lsum += t;
  }
  red[threadIdx.x] = lsum; __syncthreads();
  for (int s = 128; s > 0; s >>= 1) { if (threadIdx.x < s) red[threadIdx.x] += red[threadIdx.x + s]; __syncthreads(); }
  const float mu = red[0] * (1.0f / DD);
  __syncthreads();
  float lss = 0.0f;
#pragma unroll
  for (int i = 0; i < 3; ++i) { const float dv = v[i] - mu; lss += dv * dv; }
  red[threadIdx.x] = lss; __syncthreads();
  for (int s = 128; s > 0; s >>= 1) { if (threadIdx.x < s) red[threadIdx.x] += red[threadIdx.x + s]; __syncthreads(); }
  const float rstd = rsqrtf(red[0] * (1.0f / DD) + 1e-12f);
#pragma unroll
  for (int i = 0; i < 3; ++i) {
    const int d = threadIdx.x + i * 256;
    const float o = (v[i] - mu) * rstd * w[d] + bia[d];
    out[base + d] = o;
    if (out_bf) out_bf[base + d] = (__bf16)o;
  }
}

// ---------------------------------------------------------------------------
// Fused attention bias + softmax over one score row (b,h,i,:), f32 in, bf16 out
// ---------------------------------------------------------------------------
__global__ __launch_bounds__(256) void softmax_bias(const float* __restrict__ scores,
                                                    __bf16* __restrict__ attn,
                                                    const float* __restrict__ p,
                                                    const float* __restrict__ s,
                                                    const int* __restrict__ mask,
                                                    const float* __restrict__ compat_l,
                                                    const float* __restrict__ gamma, int l) {
  __shared__ float red[256];
  const int i  = blockIdx.x & (TT - 1);
  const int bh = blockIdx.x >> 9;
  const int h  = bh % HH, b = bh / HH;
  const float g = gamma[l];
  const float* cm = compat_l + h * 4;   // compat[l][h][k][k']
  const size_t ti = (size_t)b * TT + i;
  const float pi0 = p[ti * 2], pi1 = p[ti * 2 + 1];
  const float pc0 = pi0 * cm[0] + pi1 * cm[2];
  const float pc1 = pi0 * cm[1] + pi1 * cm[3];
  const float* row = scores + (size_t)bh * TT * TT + (size_t)i * TT;
  __bf16* orow = attn + (size_t)bh * TT * TT + (size_t)i * TT;
  float vals[2]; float lmax = -3.0e38f;
#pragma unroll
  for (int e = 0; e < 2; ++e) {
    const int j = threadIdx.x + e * 256;
    const size_t tj = (size_t)b * TT + j;
    const float bias = pc0 * p[tj * 2] + pc1 * p[tj * 2 + 1] + g * s[tj]
                     + (1.0f - (float)mask[tj]) * (-10000.0f);
    vals[e] = row[j] + bias;
    lmax = fmaxf(lmax, vals[e]);
  }
  red[threadIdx.x] = lmax; __syncthreads();
  for (int sh = 128; sh > 0; sh >>= 1) { if (threadIdx.x < sh) red[threadIdx.x] = fmaxf(red[threadIdx.x], red[threadIdx.x + sh]); __syncthreads(); }
  const float mx = red[0]; __syncthreads();
  float lsum = 0.0f;
#pragma unroll
  for (int e = 0; e < 2; ++e) { vals[e] = expf(vals[e] - mx); lsum += vals[e]; }
  red[threadIdx.x] = lsum; __syncthreads();
  for (int sh = 128; sh > 0; sh >>= 1) { if (threadIdx.x < sh) red[threadIdx.x] += red[threadIdx.x + sh]; __syncthreads(); }
  const float inv = 1.0f / red[0];
#pragma unroll
  for (int e = 0; e < 2; ++e) orow[threadIdx.x + e * 256] = (__bf16)(vals[e] * inv);
}

// ---------------------------------------------------------------------------
// Embedding gather + prototype softmax p (K=2); one block per token
// ---------------------------------------------------------------------------
__global__ __launch_bounds__(256) void embed_kernel(const int* __restrict__ ids,
                                                    const float* __restrict__ tok,
                                                    const float* __restrict__ pos,
                                                    const float* __restrict__ proto,
                                                    const float* __restrict__ ltau,
                                                    float* __restrict__ hb,
                                                    float* __restrict__ p) {
  __shared__ float r0[256], r1[256];
  const int token = blockIdx.x;
  const int t = token & (TT - 1);
  const int id = ids[token];
  float a0 = 0.0f, a1 = 0.0f;
#pragma unroll
  for (int i = 0; i < 3; ++i) {
    const int d = threadIdx.x + i * 256;
    const float vv = tok[(size_t)id * DD + d] + pos[(size_t)t * DD + d];
    hb[(size_t)token * DD + d] = vv;
    a0 += vv * proto[d];
    a1 += vv * proto[DD + d];
  }
  r0[threadIdx.x] = a0; r1[threadIdx.x] = a1; __syncthreads();
  for (int s = 128; s > 0; s >>= 1) {
    if (threadIdx.x < s) { r0[threadIdx.x] += r0[threadIdx.x + s]; r1[threadIdx.x] += r1[threadIdx.x + s]; }
    __syncthreads();
  }
  if (threadIdx.x == 0) {
    const float tau = fmaxf(expf(ltau[0]), 0.25f);
    const float l0 = r0[0] / tau, l1 = r1[0] / tau;
    const float mx = fmaxf(l0, l1);
    const float e0 = expf(l0 - mx), e1 = expf(l1 - mx);
    const float inv = 1.0f / (e0 + e1);
    p[(size_t)token * 2]     = e0 * inv;
    p[(size_t)token * 2 + 1] = e1 * inv;
  }
}

__global__ __launch_bounds__(256) void switch_kernel(const float* __restrict__ p,
                                                     float* __restrict__ s) {
  const int idx = blockIdx.x * 256 + threadIdx.x;
  if (idx >= BT) return;
  const int t = idx & (TT - 1);
  s[idx] = (t == 0) ? 0.0f
                    : 1.0f - (p[(size_t)idx * 2] * p[(size_t)(idx - 1) * 2] +
                              p[(size_t)idx * 2 + 1] * p[(size_t)(idx - 1) * 2 + 1]);
}

__global__ __launch_bounds__(256) void aug_kernel(float* __restrict__ hb,
                                                  const float* __restrict__ p,
                                                  const float* __restrict__ s,
                                                  const float* __restrict__ lang,
                                                  const float* __restrict__ sw) {
  const size_t idx = (size_t)blockIdx.x * 256 + threadIdx.x;
  const int d = (int)(idx % DD);
  const size_t token = idx / DD;
  hb[idx] += p[token * 2] * lang[d] + p[token * 2 + 1] * lang[DD + d] + s[token] * sw[d];
}

// ---------------------------------------------------------------------------
extern "C" void kernel_launch(void* const* d_in, const int* in_sizes, int n_in,
                              void* d_out, int out_size, void* d_ws, size_t ws_size,
                              hipStream_t stream) {
  (void)in_sizes; (void)n_in; (void)out_size; (void)ws_size;
  const int*   ids    = (const int*)d_in[0];
  const int*   mask   = (const int*)d_in[1];
  const float* tok    = (const float*)d_in[2];
  const float* pos    = (const float*)d_in[3];
  const float* lang   = (const float*)d_in[4];
  const float* swe    = (const float*)d_in[5];
  const float* proto  = (const float*)d_in[6];
  const float* ltau   = (const float*)d_in[7];
  const float* elnw   = (const float*)d_in[8];
  const float* elnb   = (const float*)d_in[9];
  const float* Wq     = (const float*)d_in[10];
  const float* Wk     = (const float*)d_in[11];
  const float* Wv     = (const float*)d_in[12];
  const float* Wo     = (const float*)d_in[13];
  const float* Wob    = (const float*)d_in[14];
  const float* compat = (const float*)d_in[15];
  const float* gamma  = (const float*)d_in[16];
  const float* W1     = (const float*)d_in[17];
  const float* b1     = (const float*)d_in[18];
  const float* W2     = (const float*)d_in[19];
  const float* b2     = (const float*)d_in[20];
  const float* ln1w   = (const float*)d_in[21];
  const float* ln1b   = (const float*)d_in[22];
  const float* ln2w   = (const float*)d_in[23];
  const float* ln2b   = (const float*)d_in[24];

  char* ws = (char*)d_ws;
  size_t off = 0;
  auto alloc = [&](size_t bytes) -> char* {
    char* ptr = ws + off;
    off += (bytes + 255) & ~(size_t)255;
    return ptr;
  };
  // bf16 K-major weights
  __bf16* wtq = (__bf16*)alloc((size_t)LL * DD * DD * 2);
  __bf16* wtk = (__bf16*)alloc((size_t)LL * DD * DD * 2);
  __bf16* wtv = (__bf16*)alloc((size_t)LL * DD * DD * 2);
  __bf16* wto = (__bf16*)alloc((size_t)LL * DD * DD * 2);
  __bf16* wt1 = (__bf16*)alloc((size_t)LL * DD * FF * 2);
  __bf16* wt2 = (__bf16*)alloc((size_t)LL * FF * DD * 2);
  // f32 residual-stream activations
  float* h    = (float*)alloc((size_t)BT * DD * 4);
  float* hb   = (float*)alloc((size_t)BT * DD * 4);
  float* ao   = (float*)alloc((size_t)BT * DD * 4);
  float* hmid = (float*)alloc((size_t)BT * DD * 4);
  // bf16 GEMM-operand activations
  __bf16* h_bf    = (__bf16*)alloc((size_t)BT * DD * 2);
  __bf16* hmid_bf = (__bf16*)alloc((size_t)BT * DD * 2);
  __bf16* q_bf    = (__bf16*)alloc((size_t)BT * DD * 2);
  __bf16* kt_bf   = (__bf16*)alloc((size_t)BB * HH * DHD * TT * 2);
  __bf16* v_bf    = (__bf16*)alloc((size_t)BT * DD * 2);
  __bf16* ctx_bf  = (__bf16*)alloc((size_t)BT * DD * 2);
  // small side-path
  float* p = (float*)alloc((size_t)BT * 2 * 4);
  float* s = (float*)alloc((size_t)BT * 4);
  // big unions: scores f32; attn(bf16)/ffn1(bf16) share (disjoint lifetimes)
  float*  sc      = (float*)alloc((size_t)BB * HH * TT * TT * 4);
  __bf16* attn_bf = (__bf16*)alloc((size_t)BB * HH * TT * TT * 2);
  __bf16* ffn1_bf = attn_bf;  // BT*FF*2 = 25MB < 50MB

  // Weight transpose + bf16 convert (once per call)
  dim3 tb(32, 8);
  wt_transpose<<<dim3(DD/32, DD/32, LL), tb, 0, stream>>>(Wq, wtq, DD, DD);
  wt_transpose<<<dim3(DD/32, DD/32, LL), tb, 0, stream>>>(Wk, wtk, DD, DD);
  wt_transpose<<<dim3(DD/32, DD/32, LL), tb, 0, stream>>>(Wv, wtv, DD, DD);
  wt_transpose<<<dim3(DD/32, DD/32, LL), tb, 0, stream>>>(Wo, wto, DD, DD);
  wt_transpose<<<dim3(DD/32, FF/32, LL), tb, 0, stream>>>(W1, wt1, FF, DD);
  wt_transpose<<<dim3(FF/32, DD/32, LL), tb, 0, stream>>>(W2, wt2, DD, FF);

  // Embedding stage
  embed_kernel<<<BT, 256, 0, stream>>>(ids, tok, pos, proto, ltau, hb, p);
  switch_kernel<<<BT/256, 256, 0, stream>>>(p, s);
  aug_kernel<<<(BT * DD) / 256, 256, 0, stream>>>(hb, p, s, lang, swe);
  ln_kernel<<<BT, 256, 0, stream>>>(hb, nullptr, elnw, elnb, h, h_bf);

  const dim3 blk128(128);
  const dim3 gp_d(DD/64, BT/64);        // 12 x 64
  const dim3 gp_f(FF/64, BT/64);        // 48 x 64
  const dim3 gsc(TT/64, TT/64, BB*HH);  // 8 x 8 x 96
  const dim3 gctx(1, TT/64, BB*HH);     // 1 x 8 x 96

  for (int l = 0; l < LL; ++l) {
    const __bf16* wq_ = wtq + (size_t)l * DD * DD;
    const __bf16* wk_ = wtk + (size_t)l * DD * DD;
    const __bf16* wv_ = wtv + (size_t)l * DD * DD;
    const __bf16* wo_ = wto + (size_t)l * DD * DD;
    gemm_proj<false,1><<<gp_d, blk128, 0, stream>>>(h_bf, wq_, nullptr, q_bf,  BT, DD, DD);
    gemm_proj<false,2><<<gp_d, blk128, 0, stream>>>(h_bf, wk_, nullptr, kt_bf, BT, DD, DD);
    gemm_proj<false,1><<<gp_d, blk128, 0, stream>>>(h_bf, wv_, nullptr, v_bf,  BT, DD, DD);
    scores_gemm<<<gsc, blk128, 0, stream>>>(q_bf, kt_bf, sc);
    softmax_bias<<<BB*HH*TT, 256, 0, stream>>>(sc, attn_bf, p, s, mask,
                                               compat + (size_t)l * HH * 4, gamma, l);
    ctx_gemm<<<gctx, blk128, 0, stream>>>(attn_bf, v_bf, ctx_bf);
    gemm_proj<false,0><<<gp_d, blk128, 0, stream>>>(ctx_bf, wo_, Wob + (size_t)l * DD, ao, BT, DD, DD);
    ln_kernel<<<BT, 256, 0, stream>>>(ao, h, ln1w + (size_t)l * DD, ln1b + (size_t)l * DD,
                                      hmid, hmid_bf);
    gemm_proj<true ,1><<<gp_f, blk128, 0, stream>>>(hmid_bf, wt1 + (size_t)l * DD * FF,
                                                    b1 + (size_t)l * FF, ffn1_bf, BT, FF, DD);
    gemm_proj<false,0><<<gp_d, blk128, 0, stream>>>(ffn1_bf, wt2 + (size_t)l * FF * DD,
                                                    b2 + (size_t)l * DD, ao, BT, DD, FF);
    float* hout = (l == LL - 1) ? (float*)d_out : h;
    ln_kernel<<<BT, 256, 0, stream>>>(ao, hmid, ln2w + (size_t)l * DD, ln2b + (size_t)l * DD,
                                      hout, h_bf);
  }
}
